// Actor_62010737819650
// MI455X (gfx1250) — matrix operand order repaired and checked by reading, not compiled, not run on hardware
//
#include <hip/hip_runtime.h>
#include <math.h>

typedef __attribute__((ext_vector_type(16))) __bf16 v16bf;
typedef __attribute__((ext_vector_type(8)))  float  v8f;

static __device__ __forceinline__ v8f wmma_bf16(v16bf a, v16bf b, v8f c) {
  // D = A(16x32 bf16) * B(32x16 bf16) + C(16x16 f32)
  return __builtin_amdgcn_wmma_f32_16x16x32_bf16(false, a, false, b, (short)0, c,
                                                 false, false);
}

static __device__ __forceinline__ void atomicAddGlobalF(float* p, float v) {
  __hip_atomic_fetch_add(p, v, __ATOMIC_RELAXED, __HIP_MEMORY_SCOPE_AGENT);
}

// Async global->LDS copy (CDNA5, tracked by ASYNCcnt). GV addressing form.
static __device__ __forceinline__ void async_b128_to_lds(unsigned int lds_off,
                                                         const void* gptr) {
  unsigned long long ga = (unsigned long long)(uintptr_t)gptr;
  asm volatile("global_load_async_to_lds_b128 %0, %1, off"
               :
               : "v"(lds_off), "v"(ga)
               : "memory");
}
static __device__ __forceinline__ void wait_async0() {
  asm volatile("s_wait_asynccnt 0x0" ::: "memory");
}

// ---------------------------------------------------------------------------
// workspace zeroing (float4 stores)
// ---------------------------------------------------------------------------
__global__ __launch_bounds__(256) void zero_f32(float* __restrict__ p, long long n) {
  const long long n4 = n >> 2;
  float4*         p4 = (float4*)p;
  long long       i  = (long long)blockIdx.x * blockDim.x + threadIdx.x;
  const long long st = (long long)gridDim.x * blockDim.x;
  const float4    z  = make_float4(0.0f, 0.0f, 0.0f, 0.0f);
  for (long long j = i; j < n4; j += st) p4[j] = z;
  if (i == 0)
    for (long long k = n4 << 2; k < n; ++k) p[k] = 0.0f;
}

// ---------------------------------------------------------------------------
// Fused GINE message + aggregate:
//   aggr[dst[e]] += relu(feat[src[e]] + edge_attr[e] @ We + be)
// D = feature dim (== edge-linear out dim): 32 (layer1) or 64 (layer2).
// Each wave handles 16 edges; edge-linear via v_wmma_f32_16x16x32_bf16
// (K padded 16 -> 32 with zeros). edge_attr rows streamed to LDS with
// global_load_async_to_lds_b128.
// All feature/aggr indexing is 32-bit (max N*64 = 6.4M elements) so the
// backend keeps GLOBAL (saddr+voffset) addressing, not FLAT.
// ---------------------------------------------------------------------------
template <int D>
__global__ __launch_bounds__(256)
void gine_msg(const float* __restrict__ feat, const float* __restrict__ eattr,
              const int* __restrict__ src, const int* __restrict__ dst,
              const float* __restrict__ We, const float* __restrict__ be,
              float* __restrict__ aggr, int E) {
  constexpr int NT = D / 16;
  __shared__ __bf16 wlds[16][D];      // We in bf16 (K rows 16..31 implicit zeros)
  __shared__ float  blds[D];
  __shared__ float  atile[8][16][16]; // per-wave 16 edges x 16 edge-features (f32)

  const int tid  = threadIdx.x;
  const int lane = tid & 31;
  const int wave = tid >> 5;

  for (int i = tid; i < 16 * D; i += 256) wlds[i / D][i % D] = (__bf16)We[i];
  for (int i = tid; i < D; i += 256) blds[i] = be[i];
  __syncthreads();

  // B fragments (registers, block lifetime).
  // Layout: lanes 0-15 -> K=0..15 (real), lanes 16-31 -> K=16..31 (zero pad).
  v16bf bfrag[NT];
  {
    const int  n  = lane & 15;
    const bool hi = lane >= 16;
#pragma unroll
    for (int t = 0; t < NT; ++t) {
#pragma unroll
      for (int j = 0; j < 16; ++j)
        bfrag[t][j] = hi ? (__bf16)0.0f : wlds[j][t * 16 + n];
    }
  }

  // e0 is wave-uniform: force into an SGPR so "full" becomes a scalar branch.
  const int  e0   = __builtin_amdgcn_readfirstlane(blockIdx.x * 128 + wave * 16);
  const bool full = (e0 + 16 <= E);
  const int  row  = lane & 15;
  const int  half = lane >> 4;
  const int  rb   = half * 8;  // first of the 8 C-rows this lane holds

  // Stage 16 x 16 f32 edge_attr tile into LDS (async), and overlap the
  // src/dst index fetch with it. Keep 32-bit element offsets (not pointers)
  // across the barrier so addrspace inference stays GLOBAL.
  int fo[8], ao[8];
  if (full) {
    {  // all 32 lanes: each copies 32B of one row via 2 async b128 ops
      const float*       gp = eattr + ((e0 + row) * 16 + half * 8);
      const unsigned int lo =
          (unsigned int)(uintptr_t)(void*)&atile[wave][row][half * 8];
      async_b128_to_lds(lo, gp);
      async_b128_to_lds(lo + 16, gp + 4);
    }
    const int4 a = *(const int4*)(src + e0 + rb);
    const int4 b = *(const int4*)(src + e0 + rb + 4);
    const int4 c = *(const int4*)(dst + e0 + rb);
    const int4 d = *(const int4*)(dst + e0 + rb + 4);
    fo[0] = a.x * D; fo[1] = a.y * D; fo[2] = a.z * D; fo[3] = a.w * D;
    fo[4] = b.x * D; fo[5] = b.y * D; fo[6] = b.z * D; fo[7] = b.w * D;
    ao[0] = c.x * D; ao[1] = c.y * D; ao[2] = c.z * D; ao[3] = c.w * D;
    ao[4] = d.x * D; ao[5] = d.y * D; ao[6] = d.z * D; ao[7] = d.w * D;
  } else if (lane < 16) {
    const int e = e0 + row;
    if (e < E) {
      const float* p = eattr + e * 16;
#pragma unroll
      for (int j = 0; j < 16; ++j) atile[wave][row][j] = p[j];
    } else {
#pragma unroll
      for (int j = 0; j < 16; ++j) atile[wave][row][j] = 0.0f;
    }
  }
  wait_async0();
  __syncthreads();

  // A fragment (contiguous order k = 16*half + j, as in bfrag); K=16..31 zero (ED == 16).
  v16bf afrag;
  {
    const int kb = half ? 8 : 0;
#pragma unroll
    for (int j = 0; j < 8; ++j) {
      afrag[j]     = half ? (__bf16)0.0f : (__bf16)atile[wave][row][j];
      afrag[8 + j] = half ? (__bf16)0.0f : (__bf16)atile[wave][row][8 + j];
    }
  }

  // All WMMA accumulators first (uniform EXEC), epilogue after.
  v8f cts[NT];
#pragma unroll
  for (int t = 0; t < NT; ++t) {
    v8f c  = {};
    cts[t] = wmma_bf16(afrag, bfrag[t], c);
  }

  if (full) {
#pragma unroll
    for (int t = 0; t < NT; ++t) {
      const int   col  = t * 16 + row;
      const float bias = blds[col];
      // phase 1: issue all 8 gathers
      float fv[8];
#pragma unroll
      for (int r = 0; r < 8; ++r) fv[r] = feat[fo[r] + col];
      // phase 2: combine + scatter-add
#pragma unroll
      for (int r = 0; r < 8; ++r) {
        float v = fv[r] + cts[t][r] + bias;
        v = v > 0.0f ? v : 0.0f;
        atomicAddGlobalF(&aggr[ao[r] + col], v);
      }
    }
  } else {
#pragma unroll
    for (int t = 0; t < NT; ++t) {
      const int   col  = t * 16 + row;
      const float bias = blds[col];
#pragma unroll
      for (int r = 0; r < 8; ++r) {
        const int e = e0 + rb + r;
        if (e < E) {
          const int s = src[e];
          const int d = dst[e];
          float v = feat[s * D + col] + cts[t][r] + bias;
          v = v > 0.0f ? v : 0.0f;
          atomicAddGlobalF(&aggr[d * D + col], v);
        }
      }
    }
  }
}

// ---------------------------------------------------------------------------
// GINE node update MLP:
//   h = relu((x + aggr) @ Wa + ba) @ Wb + bb
// KIN = input dim (32 or 64), hidden/out = 64. Each wave owns 16 nodes.
// FINAL: accumulate column sums into gsum (global mean pool) instead of
// storing h.
// ---------------------------------------------------------------------------
template <int KIN, bool FINAL>
__global__ __launch_bounds__(256)
void gine_node(const float* __restrict__ xin, const float* __restrict__ aggr,
               const float* __restrict__ Wa, const float* __restrict__ ba,
               const float* __restrict__ Wb, const float* __restrict__ bb,
               float* __restrict__ hout, float* __restrict__ gsum, int N) {
  __shared__ __bf16 w1[KIN][64];
  __shared__ __bf16 w2[64][64];
  __shared__ float  bav[64];
  __shared__ float  bbv[64];
  __shared__ __bf16 itile[8][16][KIN];
  __shared__ __bf16 mtile[8][16][64];
  __shared__ float  red[64];

  const int tid  = threadIdx.x;
  const int lane = tid & 31;
  const int wave = tid >> 5;

  for (int i = tid; i < KIN * 64; i += 256) w1[i / 64][i % 64] = (__bf16)Wa[i];
  for (int i = tid; i < 64 * 64; i += 256) w2[i / 64][i % 64] = (__bf16)Wb[i];
  if (tid < 64) {
    bav[tid] = ba[tid];
    bbv[tid] = bb[tid];
    red[tid] = 0.0f;
  }
  __syncthreads();

  const int  n0   = __builtin_amdgcn_readfirstlane(blockIdx.x * 128 + wave * 16);
  const bool full = (n0 + 16 <= N);
  const int  row  = lane & 15;
  const int  half = lane >> 4;

  // Stage (x + aggr) tile; each half-wave handles half of K.
  {
    constexpr int KH = KIN / 2;
    const int     n  = n0 + row;
    if (n < N) {
      const int base = n * KIN + half * KH;
#pragma unroll
      for (int k = 0; k < KH; ++k)
        itile[wave][row][half * KH + k] = (__bf16)(xin[base + k] + aggr[base + k]);
    } else {
#pragma unroll
      for (int k = 0; k < KH; ++k)
        itile[wave][row][half * KH + k] = (__bf16)0.0f;
    }
  }
  __syncthreads();

  const int kb8  = half ? 8 : 0;
  const int kb16 = half ? 16 : 0;
  const int rb   = half * 8;

  v16bf a0, a1;
#pragma unroll
  for (int j = 0; j < 8; ++j) {
    a0[j]     = itile[wave][row][kb8 + j];
    a0[8 + j] = itile[wave][row][16 + kb8 + j];
  }
  if (KIN == 64) {
#pragma unroll
    for (int j = 0; j < 8; ++j) {
      a1[j]     = itile[wave][row][32 + kb8 + j];
      a1[8 + j] = itile[wave][row][48 + kb8 + j];
    }
  }

  // GEMM1 + bias + ReLU -> mtile (bf16)
#pragma unroll
  for (int t = 0; t < 4; ++t) {
    v16bf b0;
#pragma unroll
    for (int j = 0; j < 16; ++j) b0[j] = w1[kb8 + (j & 7) + (j & 8) * 2][t * 16 + row];
    v8f c = {};
    c = wmma_bf16(a0, b0, c);
    if (KIN == 64) {
      v16bf b1;
#pragma unroll
      for (int j = 0; j < 16; ++j) b1[j] = w1[32 + kb8 + (j & 7) + (j & 8) * 2][t * 16 + row];
      c = wmma_bf16(a1, b1, c);
    }
    const int   col  = t * 16 + row;
    const float bias = bav[col];
#pragma unroll
    for (int r = 0; r < 8; ++r) {
      float v = c[r] + bias;
      v = v > 0.0f ? v : 0.0f;
      mtile[wave][rb + r][col] = (__bf16)v;
    }
  }
  __syncthreads();

  v16bf h0, h1;
#pragma unroll
  for (int j = 0; j < 8; ++j) {
    h0[j]     = mtile[wave][row][kb8 + j];
    h0[8 + j] = mtile[wave][row][16 + kb8 + j];
    h1[j]     = mtile[wave][row][32 + kb8 + j];
    h1[8 + j] = mtile[wave][row][48 + kb8 + j];
  }

  // GEMM2 accumulators first, then uniform-branch epilogue.
  v8f cts[4];
#pragma unroll
  for (int t = 0; t < 4; ++t) {
    v16bf b0, b1;
#pragma unroll
    for (int j = 0; j < 16; ++j) {
      b0[j] = w2[kb8 + (j & 7) + (j & 8) * 2][t * 16 + row];
      b1[j] = w2[32 + kb8 + (j & 7) + (j & 8) * 2][t * 16 + row];
    }
    v8f c = {};
    c      = wmma_bf16(h0, b0, c);
    cts[t] = wmma_bf16(h1, b1, c);
  }

  if (FINAL) {
    float acc[4] = {0.0f, 0.0f, 0.0f, 0.0f};
    if (full) {
#pragma unroll
      for (int t = 0; t < 4; ++t) {
        const float bias = bbv[t * 16 + row];
#pragma unroll
        for (int r = 0; r < 8; ++r) acc[t] += cts[t][r] + bias;
      }
    } else {
#pragma unroll
      for (int t = 0; t < 4; ++t) {
        const float bias = bbv[t * 16 + row];
#pragma unroll
        for (int r = 0; r < 8; ++r)
          if (n0 + rb + r < N) acc[t] += cts[t][r] + bias;
      }
    }
#pragma unroll
    for (int t = 0; t < 4; ++t) atomicAdd(&red[t * 16 + row], acc[t]);
    __syncthreads();
    if (tid < 64) atomicAddGlobalF(&gsum[tid], red[tid]);
  } else {
    if (full) {
      int ho[8];
#pragma unroll
      for (int r = 0; r < 8; ++r) ho[r] = (n0 + rb + r) * 64;
#pragma unroll
      for (int t = 0; t < 4; ++t) {
        const int   col  = t * 16 + row;
        const float bias = bbv[col];
#pragma unroll
        for (int r = 0; r < 8; ++r) hout[ho[r] + col] = cts[t][r] + bias;
      }
    } else {
#pragma unroll
      for (int t = 0; t < 4; ++t) {
        const int   col  = t * 16 + row;
        const float bias = bbv[col];
#pragma unroll
        for (int r = 0; r < 8; ++r) {
          const int n = n0 + rb + r;
          if (n < N) hout[n * 64 + col] = cts[t][r] + bias;
        }
      }
    }
  }
}

// ---------------------------------------------------------------------------
// Head MLP (tiny, one block).
// ---------------------------------------------------------------------------
__global__ __launch_bounds__(256)
void head_kernel(const float* __restrict__ gsum, float invN,
                 const float* __restrict__ Wn1, const float* __restrict__ bn1,
                 const float* __restrict__ Wn2, const float* __restrict__ bn2,
                 const float* __restrict__ Ws1, const float* __restrict__ bs1,
                 const float* __restrict__ Ws2, const float* __restrict__ bs2,
                 const float* __restrict__ Wt1, const float* __restrict__ bt1,
                 const float* __restrict__ Wt2, const float* __restrict__ bt2,
                 float* __restrict__ out) {
  __shared__ float g[64], z[256], z2[128], hs[64], ht[64];
  const int t = threadIdx.x;
  if (t < 64) g[t] = gsum[t] * invN;
  __syncthreads();
  {
    float a = bn1[t];
    for (int c = 0; c < 64; ++c) a += g[c] * Wn1[c * 256 + t];
    z[t] = a > 0.0f ? a : 0.0f;
  }
  __syncthreads();
  if (t < 128) {
    float a = bn2[t];
    for (int k = 0; k < 256; ++k) a += z[k] * Wn2[k * 128 + t];
    z2[t] = a > 0.0f ? a : 0.0f;
  }
  __syncthreads();
  if (t < 64) {
    float a = bs1[t];
    float b = bt1[t];
    for (int k = 0; k < 128; ++k) {
      a += z2[k] * Ws1[k * 64 + t];
      b += z2[k] * Wt1[k * 64 + t];
    }
    hs[t] = a > 0.0f ? a : 0.0f;
    ht[t] = b > 0.0f ? b : 0.0f;
  }
  __syncthreads();
  if (t < 64) {
    float a = bs2[t];
    for (int k = 0; k < 64; ++k) a += hs[k] * Ws2[k * 64 + t];
    out[t] = 1.0f / (1.0f + expf(-a));
  }
  if (t >= 64 && t < 96) {
    const int i = t - 64;
    float a = bt2[i];
    for (int k = 0; k < 64; ++k) a += ht[k] * Wt2[k * 32 + i];
    out[64 + i] = 1.0f / (1.0f + expf(-a));
  }
}

// ---------------------------------------------------------------------------
extern "C" void kernel_launch(void* const* d_in, const int* in_sizes, int n_in,
                              void* d_out, int out_size, void* d_ws, size_t ws_size,
                              hipStream_t stream) {
  const int N = in_sizes[0] / 32;
  const int E = in_sizes[1] / 16;

  const float* x   = (const float*)d_in[0];
  const float* ea  = (const float*)d_in[1];
  const int*   ei  = (const int*)d_in[2];
  const int*   src = ei;
  const int*   dst = ei + E;
  const float* We1 = (const float*)d_in[3];  const float* be1 = (const float*)d_in[4];
  const float* W1a = (const float*)d_in[5];  const float* b1a = (const float*)d_in[6];
  const float* W1b = (const float*)d_in[7];  const float* b1b = (const float*)d_in[8];
  const float* We2 = (const float*)d_in[9];  const float* be2 = (const float*)d_in[10];
  const float* W2a = (const float*)d_in[11]; const float* b2a = (const float*)d_in[12];
  const float* W2b = (const float*)d_in[13]; const float* b2b = (const float*)d_in[14];
  const float* Wn1 = (const float*)d_in[15]; const float* bn1 = (const float*)d_in[16];
  const float* Wn2 = (const float*)d_in[17]; const float* bn2 = (const float*)d_in[18];
  const float* Ws1 = (const float*)d_in[19]; const float* bs1 = (const float*)d_in[20];
  const float* Ws2 = (const float*)d_in[21]; const float* bs2 = (const float*)d_in[22];
  const float* Wt1 = (const float*)d_in[23]; const float* bt1 = (const float*)d_in[24];
  const float* Wt2 = (const float*)d_in[25]; const float* bt2 = (const float*)d_in[26];

  // workspace layout (f32): aggr1[N*32] | aggr2[N*64] | gsum[64] | h[N*64]
  float* ws    = (float*)d_ws;
  float* aggr1 = ws;
  float* aggr2 = aggr1 + (size_t)N * 32;
  float* gsum  = aggr2 + (size_t)N * 64;
  float* h     = gsum + 64;

  const long long zn = (long long)N * 96 + 64;
  zero_f32<<<2048, 256, 0, stream>>>(ws, zn);

  const int egrid = (E + 127) / 128;
  const int ngrid = (N + 127) / 128;

  gine_msg<32><<<egrid, 256, 0, stream>>>(x, ea, src, dst, We1, be1, aggr1, E);
  gine_node<32, false><<<ngrid, 256, 0, stream>>>(x, aggr1, W1a, b1a, W1b, b1b, h, gsum, N);
  gine_msg<64><<<egrid, 256, 0, stream>>>(h, ea, src, dst, We2, be2, aggr2, E);
  gine_node<64, true><<<ngrid, 256, 0, stream>>>(h, aggr2, W2a, b2a, W2b, b2b, h, gsum, N);
  head_kernel<<<1, 256, 0, stream>>>(gsum, 1.0f / (float)N, Wn1, bn1, Wn2, bn2,
                                     Ws1, bs1, Ws2, bs2, Wt1, bt1, Wt2, bt2,
                                     (float*)d_out);
}
